// MultiHeadSelfAttention_88167088652862
// MI455X (gfx1250) — matrix operand order repaired and checked by reading, not compiled, not run on hardware
//
#include <hip/hip_runtime.h>
#include <hip/hip_bf16.h>

typedef __attribute__((ext_vector_type(16))) _Float16 v16h;
typedef __attribute__((ext_vector_type(4)))  _Float16 v4h;
typedef __attribute__((ext_vector_type(8)))  float    v8f;
typedef __attribute__((ext_vector_type(4)))  int      v4i;

#define WMMA_F32_F16(a, b, c) \
    __builtin_amdgcn_wmma_f32_16x16x32_f16(false, (a), false, (b), (short)0, (c), false, false)

#define GLOBAL_AS __attribute__((address_space(1)))
#define LDS_AS    __attribute__((address_space(3)))

// ---------------------------------------------------------------------------
// Async global->LDS 16-byte copy (CDNA5 GLOBAL_LOAD_ASYNC_TO_LDS_B128,
// tracked with ASYNCcnt). Falls back to a plain copy if the builtin is absent.
// ---------------------------------------------------------------------------
#if __has_builtin(__builtin_amdgcn_global_load_async_to_lds_b128)
#define HAVE_ASYNC_LDS 1
__device__ __forceinline__ void async_copy16(const _Float16* g, _Float16* l) {
    __builtin_amdgcn_global_load_async_to_lds_b128(
        (GLOBAL_AS v4i*)g, (LDS_AS v4i*)l, 0, 0);
}
#else
#define HAVE_ASYNC_LDS 0
__device__ __forceinline__ void async_copy16(const _Float16* g, _Float16* l) {
    #pragma unroll
    for (int i = 0; i < 8; ++i) l[i] = g[i];
}
#endif

template <int N>
__device__ __forceinline__ void wait_async() {
#if HAVE_ASYNC_LDS
#if __has_builtin(__builtin_amdgcn_s_wait_asynccnt)
    __builtin_amdgcn_s_wait_asynccnt(N);
#else
    asm volatile("s_wait_asynccnt %0" :: "i"(N) : "memory");
#endif
#endif
}

// ---------------------------------------------------------------------------
// DPP16 butterfly reductions across each 16-lane half (no LDS, no dscnt).
// ---------------------------------------------------------------------------
template <int CTRL>
__device__ __forceinline__ float dpp_movf(float x) {
    int i = __builtin_bit_cast(int, x);
    int r = __builtin_amdgcn_update_dpp(i, i, CTRL, 0xF, 0xF, true);
    return __builtin_bit_cast(float, r);
}
__device__ __forceinline__ float redmax16(float x) {
    x = fmaxf(x, dpp_movf<0xB1>(x));    // quad_perm xor1
    x = fmaxf(x, dpp_movf<0x4E>(x));    // quad_perm xor2
    x = fmaxf(x, dpp_movf<0x141>(x));   // row_half_mirror (xor7 within 8)
    x = fmaxf(x, dpp_movf<0x140>(x));   // row_mirror (xor15 within 16)
    return x;
}
__device__ __forceinline__ float redsum16(float x) {
    x += dpp_movf<0xB1>(x);
    x += dpp_movf<0x4E>(x);
    x += dpp_movf<0x141>(x);
    x += dpp_movf<0x140>(x);
    return x;
}

// ---------------------------------------------------------------------------
// Elementwise f32 -> f16 conversion (vectorized 4-wide). n divisible by 1024.
// ---------------------------------------------------------------------------
__global__ __launch_bounds__(256)
void cvt_f32_f16(const float* __restrict__ src, _Float16* __restrict__ dst, int n4) {
    int i = blockIdx.x * 256 + threadIdx.x;
    if (i < n4) {
        float4 v = ((const float4*)src)[i];
        v4h o;
        o[0] = (_Float16)v.x; o[1] = (_Float16)v.y;
        o[2] = (_Float16)v.z; o[3] = (_Float16)v.w;
        ((v4h*)dst)[i] = o;
    }
}

// ---------------------------------------------------------------------------
// Tiled GEMM with bias: out[M,N] = A[M,K] * W[K,N] + bias[N], A/W in f16.
// Block tile 128x64, BK=32; 8 waves = 4(M) x 2(N), 32x32 per wave.
// Double-buffered async LDS staging: tile t+1 in flight behind tile t's WMMAs.
// ---------------------------------------------------------------------------
template <typename TOut>
__global__ __launch_bounds__(256)
void gemm_bias_f16wmma(const _Float16* __restrict__ A, const _Float16* __restrict__ W,
                       const float* __restrict__ bias, TOut* __restrict__ out,
                       int M, int N, int K) {
    constexpr int BM = 128, BN = 64, BK = 32;
    __shared__ _Float16 Xs[2][BM][BK + 8];   // row stride 80 B (16B aligned)
    __shared__ _Float16 Ws[2][BK][BN + 8];   // row stride 144 B (16B aligned)

    const int tid   = threadIdx.x;
    const int lane  = tid & 31;
    const int wave  = tid >> 5;
    const int laneM = lane & 15;
    const int hi    = lane >> 4;          // 0 = lanes 0-15, 1 = lanes 16-31

    const int m0 = blockIdx.y * BM;
    const int n0 = blockIdx.x * BN;
    const int mQ = wave & 3;              // 32-row group
    const int nH = wave >> 2;             // 32-col group

    const int arow  = tid >> 1;           // A tile: 2 threads/row, 16 elems each
    const int acol  = (tid & 1) * 16;
    const int wrow  = tid >> 3;           // W tile: 8 threads/row, 8 elems each
    const int wcol  = (tid & 7) * 8;

    v8f acc[2][2] = {};

    auto issue = [&](int kb, int buf) {
        async_copy16(A + (size_t)(m0 + arow) * K + kb + acol,     &Xs[buf][arow][acol]);
        async_copy16(A + (size_t)(m0 + arow) * K + kb + acol + 8, &Xs[buf][arow][acol + 8]);
        async_copy16(W + (size_t)(kb + wrow) * N + n0 + wcol,     &Ws[buf][wrow][wcol]);
    };
    auto compute = [&](int buf) {
        // A fragments (16x32 f16, ISA layout): row = lane&15,
        // k = (f&7) + (f>=8 ? 16 : 0) + (hiHalf ? 8 : 0)
        v16h a[2], b[2];
        #pragma unroll
        for (int s = 0; s < 2; ++s)
            #pragma unroll
            for (int f = 0; f < 16; ++f) {
                int kk = (f & 7) + ((f & 8) ? 16 : 0) + (hi ? 8 : 0);
                a[s][f] = Xs[buf][mQ * 32 + s * 16 + laneM][kk];
            }
        // B fragments (32x16): k = (f&7) + (f>=8 ? 16 : 0) + (hiHalf ? 8 : 0), col = lane&15
        #pragma unroll
        for (int s = 0; s < 2; ++s)
            #pragma unroll
            for (int f = 0; f < 16; ++f) {
                int kk = (f & 7) + ((f & 8) ? 16 : 0) + (hi ? 8 : 0);
                b[s][f] = Ws[buf][kk][nH * 32 + s * 16 + laneM];
            }
        #pragma unroll
        for (int i = 0; i < 2; ++i)
            #pragma unroll
            for (int j = 0; j < 2; ++j)
                acc[i][j] = WMMA_F32_F16(a[i], b[j], acc[i][j]);
    };

    issue(0, 0);
    int buf = 0;
    for (int kb = 0; kb + BK < K; kb += BK, buf ^= 1) {
        issue(kb + BK, buf ^ 1);          // prefetch next tile
        wait_async<3>();                  // retire current tile's 3 copies
        __syncthreads();
        compute(buf);
        __syncthreads();
    }
    wait_async<0>();
    __syncthreads();
    compute(buf);

    // Epilogue: C/D layout row = r + hi*8, col = lane&15.
    #pragma unroll
    for (int i = 0; i < 2; ++i)
        #pragma unroll
        for (int j = 0; j < 2; ++j)
            #pragma unroll
            for (int r = 0; r < 8; ++r) {
                int m = m0 + mQ * 32 + i * 16 + r + hi * 8;
                int n = n0 + nH * 32 + j * 16 + laneM;
                out[(size_t)m * N + n] = (TOut)(acc[i][j][r] + bias[n]);
            }
}

// ---------------------------------------------------------------------------
// Flash attention over f16 Q/K/V stored as [B, S, H, d] (d = 64).
// Grid: (S/128, B*H). 256 threads = 8 waves; each wave owns 16 query rows
// and streams all S keys in 32-key tiles with online softmax (DPP reductions).
// Double-buffered async K/V staging.
// ---------------------------------------------------------------------------
__global__ __launch_bounds__(256)
void flash_attn_f16wmma(const _Float16* __restrict__ Qh,
                        const _Float16* __restrict__ Kh,
                        const _Float16* __restrict__ Vh,
                        _Float16* __restrict__ Ah, int B, int S) {
    constexpr int H = 16, D = 64, KT = 32;
    __shared__ _Float16 Ks[2][KT][D + 8];    // row stride 144 B
    __shared__ _Float16 Vs[2][KT][D + 8];
    __shared__ _Float16 Ps[8][16][KT + 8];

    const int tid   = threadIdx.x;
    const int lane  = tid & 31;
    const int wave  = tid >> 5;
    const int laneM = lane & 15;
    const int hi    = lane >> 4;

    const int bh = blockIdx.y;
    const int b  = bh / H;
    const int h  = bh % H;
    const int qBase = blockIdx.x * 128 + wave * 16;
    const float scale = 0.125f;   // 1/sqrt(64)

    // Load Q A-fragments for this wave's 16 rows (d split 0..31 / 32..63).
    v16h aq0, aq1;
    {
        size_t qrow = ((size_t)(b * S + qBase + laneM) * H + h) * D;
        #pragma unroll
        for (int f = 0; f < 16; ++f) {
            int dd = (f & 7) + ((f & 8) ? 16 : 0) + (hi ? 8 : 0);
            aq0[f] = Qh[qrow + dd];
            aq1[f] = Qh[qrow + 32 + dd];
        }
    }

    v8f o0 = {}, o1 = {}, o2 = {}, o3 = {};
    float rowm[8], rowl[8];
    #pragma unroll
    for (int r = 0; r < 8; ++r) { rowm[r] = -1e30f; rowl[r] = 0.0f; }

    const int krow  = tid >> 3;           // 32 rows, 8 threads/row
    const int kcol  = (tid & 7) * 8;      // 16 B per thread per matrix

    auto issueKV = [&](int kb, int buf) {
        size_t base = ((size_t)(b * S + kb + krow) * H + h) * D + kcol;
        async_copy16(Kh + base, &Ks[buf][krow][kcol]);
        async_copy16(Vh + base, &Vs[buf][krow][kcol]);
    };

    auto computeTile = [&](int buf) {
        // Scores for two 16-key subtiles: S = Q * K^T (K-dim = d = 64).
        v8f sc[2];
        #pragma unroll
        for (int j = 0; j < 2; ++j) {
            v16h bk0, bk1;
            #pragma unroll
            for (int f = 0; f < 16; ++f) {
                int dd = (f & 7) + ((f & 8) ? 16 : 0) + (hi ? 8 : 0); // B k-index
                bk0[f] = Ks[buf][j * 16 + laneM][dd];       // K[key][dd]
                bk1[f] = Ks[buf][j * 16 + laneM][32 + dd];
            }
            v8f s = {};
            s = WMMA_F32_F16(aq0, bk0, s);
            s = WMMA_F32_F16(aq1, bk1, s);
            sc[j] = s;
        }

        // Online softmax per row (row r lives in lanes {hi*16 .. hi*16+15}).
        float p0[8], p1[8];
        #pragma unroll
        for (int r = 0; r < 8; ++r) {
            float s0 = sc[0][r] * scale;
            float s1 = sc[1][r] * scale;
            float mx = redmax16(fmaxf(s0, s1));
            float mnew = fmaxf(rowm[r], mx);
            float corr = __expf(rowm[r] - mnew);
            float e0 = __expf(s0 - mnew);
            float e1 = __expf(s1 - mnew);
            float ps = redsum16(e0 + e1);
            rowl[r] = rowl[r] * corr + ps;
            rowm[r] = mnew;
            o0[r] *= corr; o1[r] *= corr; o2[r] *= corr; o3[r] *= corr;
            p0[r] = e0; p1[r] = e1;
        }

        // Re-layout P (C layout -> A layout) through per-wave LDS scratch.
        #pragma unroll
        for (int r = 0; r < 8; ++r) {
            Ps[wave][r + hi * 8][laneM]      = (_Float16)p0[r];
            Ps[wave][r + hi * 8][16 + laneM] = (_Float16)p1[r];
        }
        v16h ap;
        #pragma unroll
        for (int f = 0; f < 16; ++f) {
            int kk = (f & 7) + ((f & 8) ? 16 : 0) + (hi ? 8 : 0);
            ap[f] = Ps[wave][laneM][kk];   // DS ops in-order within wave
        }

        // O += P * V  (K-dim = 32 keys, N = 16-wide d chunks).
        #pragma unroll
        for (int j = 0; j < 4; ++j) {
            v16h bv;
            #pragma unroll
            for (int f = 0; f < 16; ++f) {
                int kk = (f & 7) + ((f & 8) ? 16 : 0) + (hi ? 8 : 0); // key index in tile
                bv[f] = Vs[buf][kk][j * 16 + laneM];
            }
            if (j == 0) o0 = WMMA_F32_F16(ap, bv, o0);
            if (j == 1) o1 = WMMA_F32_F16(ap, bv, o1);
            if (j == 2) o2 = WMMA_F32_F16(ap, bv, o2);
            if (j == 3) o3 = WMMA_F32_F16(ap, bv, o3);
        }
    };

    issueKV(0, 0);
    int buf = 0;
    for (int kb = 0; kb + KT < S; kb += KT, buf ^= 1) {
        issueKV(kb + KT, buf ^ 1);        // prefetch next K/V tile
        wait_async<2>();                  // retire current tile's 2 copies
        __syncthreads();
        computeTile(buf);
        __syncthreads();
    }
    wait_async<0>();
    __syncthreads();
    computeTile(buf);

    // Normalize and store attended as f16 [B,S,H,d].
    #pragma unroll
    for (int r = 0; r < 8; ++r) {
        int q = qBase + r + hi * 8;
        float inv = 1.0f / rowl[r];
        size_t base = ((size_t)(b * S + q) * H + h) * D;
        Ah[base +  0 + laneM] = (_Float16)(o0[r] * inv);
        Ah[base + 16 + laneM] = (_Float16)(o1[r] * inv);
        Ah[base + 32 + laneM] = (_Float16)(o2[r] * inv);
        Ah[base + 48 + laneM] = (_Float16)(o3[r] * inv);
    }
}

// ---------------------------------------------------------------------------
extern "C" void kernel_launch(void* const* d_in, const int* in_sizes, int n_in,
                              void* d_out, int out_size, void* d_ws, size_t ws_size,
                              hipStream_t stream) {
    (void)in_sizes; (void)n_in; (void)out_size; (void)ws_size;
    const int B = 4, S = 2048, Dm = 1024;
    const int M = B * S;                 // 8192

    const float* x  = (const float*)d_in[0];
    const float* Wq = (const float*)d_in[1];
    const float* bq = (const float*)d_in[2];
    const float* Wk = (const float*)d_in[3];
    const float* bk = (const float*)d_in[4];
    const float* Wv = (const float*)d_in[5];
    const float* bv = (const float*)d_in[6];
    const float* Wo = (const float*)d_in[7];
    const float* bo = (const float*)d_in[8];
    float* out = (float*)d_out;

    // Workspace: f16 Xh[M*Dm], Wq/Wk/Wv/Wo h[Dm*Dm], Qh/Kh/Vh/Ah[M*Dm].
    const size_t me = (size_t)M * Dm;     // 8 Mi elems
    const size_t we = (size_t)Dm * Dm;    // 1 Mi elems
    _Float16* Xh  = (_Float16*)d_ws;
    _Float16* Wqh = Xh  + me;
    _Float16* Wkh = Wqh + we;
    _Float16* Wvh = Wkh + we;
    _Float16* Woh = Wvh + we;
    _Float16* Qh  = Woh + we;
    _Float16* Kh  = Qh  + me;
    _Float16* Vh  = Kh  + me;
    _Float16* Ah  = Vh  + me;

    dim3 blk(256);

    // Pre-convert activations and weights to f16 (one cheap pass).
    {
        int n4x = (M * Dm) / 4, n4w = (Dm * Dm) / 4;
        cvt_f32_f16<<<dim3(n4x / 256), blk, 0, stream>>>(x,  Xh,  n4x);
        cvt_f32_f16<<<dim3(n4w / 256), blk, 0, stream>>>(Wq, Wqh, n4w);
        cvt_f32_f16<<<dim3(n4w / 256), blk, 0, stream>>>(Wk, Wkh, n4w);
        cvt_f32_f16<<<dim3(n4w / 256), blk, 0, stream>>>(Wv, Wvh, n4w);
        cvt_f32_f16<<<dim3(n4w / 256), blk, 0, stream>>>(Wo, Woh, n4w);
    }

    dim3 gGemm(Dm / 64, M / 128);        // (16, 64)
    gemm_bias_f16wmma<_Float16><<<gGemm, blk, 0, stream>>>(Xh, Wqh, bq, Qh, M, Dm, Dm);
    gemm_bias_f16wmma<_Float16><<<gGemm, blk, 0, stream>>>(Xh, Wkh, bk, Kh, M, Dm, Dm);
    gemm_bias_f16wmma<_Float16><<<gGemm, blk, 0, stream>>>(Xh, Wvh, bv, Vh, M, Dm, Dm);

    dim3 gAttn(S / 128, B * 16);         // (16, 64)
    flash_attn_f16wmma<<<gAttn, blk, 0, stream>>>(Qh, Kh, Vh, Ah, B, S);

    gemm_bias_f16wmma<float><<<gGemm, blk, 0, stream>>>(Ah, Woh, bo, out, M, Dm, Dm);
}
